// EquivariantDiffusionModel_65008624992701
// MI455X (gfx1250) — compile-verified, matrix operand order checked by reading, not run on hardware
//
#include <hip/hip_runtime.h>
#include <hip/hip_bf16.h>
#include <math.h>

// Problem constants (match reference)
#define BB 32
#define NN 64
#define EE 4096            // N*N
#define HH 256
#define LL 9
#define ROWS  (BB*NN)      // 2048 node rows
#define EROWS (BB*EE)      // 131072 edge rows

typedef __attribute__((ext_vector_type(16))) __bf16 v16bf;
typedef __attribute__((ext_vector_type(8)))  float  v8f;

union AFrag { v16bf v; unsigned int u[8]; };

__device__ __forceinline__ unsigned short f2bf(float f){
  unsigned int u = __float_as_uint(f);
  u += 0x7fffu + ((u >> 16) & 1u);          // round-to-nearest-even
  return (unsigned short)(u >> 16);
}
__device__ __forceinline__ float bf2f(unsigned short h){
  return __uint_as_float(((unsigned int)h) << 16);
}
__device__ __forceinline__ float silu_f(float x){ return x / (1.f + __expf(-x)); }
__device__ __forceinline__ float sigm_f(float x){ return 1.f / (1.f + __expf(-x)); }

// ---------------------------------------------------------------------------
// Weight packer: f32 row-major (K x 256) -> WMMA B-fragment layout
// Bp layout: [nTile=16][kt=K/32][lane=32][q=8] uint32 (bf16 pair, K=(k,k+1))
// lane<16: columns n=lane, K-halves {0..7,16..23}; lane>=16: K {8..15,24..31}
// ---------------------------------------------------------------------------
__global__ void pack_w_k(const float* __restrict__ W, unsigned int* __restrict__ Bp, int K)
{
  int id = blockIdx.x * 256 + threadIdx.x;
  int total = 16 * (K >> 5) * 256;
  if (id >= total) return;
  int q    = id & 7;
  int lane = (id >> 3) & 31;
  int rem  = id >> 8;                 // nT*(K/32) + kt
  int ks   = K >> 5;
  int kt   = rem % ks;
  int nT   = rem / ks;
  int lc = lane & 15, hi = lane >> 4;
  int kbase = 2*(q & 3) + (hi << 3) + ((q >> 2) << 4);
  int k0 = (kt << 5) + kbase;
  int n  = (nT << 4) + lc;
  unsigned int lo  = f2bf(W[(size_t)k0      * 256 + n]);
  unsigned int hi2 = f2bf(W[(size_t)(k0+1)  * 256 + n]);
  Bp[id] = lo | (hi2 << 16);
}

// ---------------------------------------------------------------------------
// Generic WMMA GEMM: out(Mx256) = act(A(MxK,bf16) @ Wpacked + bias + addsrc)
// Block = 256 threads = 8 waves; tile 64(M) x 256(N); wave strip 64x32.
// A tile double-buffered through LDS so the global load for step kt+1 is
// issued before the WMMAs of step kt (hides HBM/L2 latency); one barrier
// per K-step instead of two.
// ---------------------------------------------------------------------------
__global__ __launch_bounds__(256) void wmma_gemm_k(
    const unsigned short* __restrict__ A,   // M x K bf16 row-major (lda=K)
    const unsigned int*   __restrict__ Bp,  // packed weights
    const float* __restrict__ bias,         // 256 or null
    const float* __restrict__ addsrc,       // M x 256 f32 residual or null
    unsigned short* __restrict__ outb,      // M x 256 bf16 or null
    float* __restrict__ outf,               // M x 256 f32 or null
    int K, int act)
{
  __shared__ __align__(16) unsigned int Asm[2][64 * 16];  // 2 x 4KB A tiles
  const int tid  = threadIdx.x;
  const int w    = tid >> 5;
  const int lane = tid & 31;
  const int lr   = lane & 15;
  const int hi   = lane >> 4;
  const long row0 = (long)blockIdx.x * 64;
  const int ksteps = K >> 5;

  v8f zero = {0.f,0.f,0.f,0.f,0.f,0.f,0.f,0.f};
  v8f acc[4][2];
  #pragma unroll
  for (int mt=0; mt<4; ++mt) { acc[mt][0] = zero; acc[mt][1] = zero; }

  const int arow = tid >> 2;           // A-tile staging: row 0..63
  const int aq0  = (tid & 3) << 2;     // dword q 0,4,8,12
  const unsigned short* abase = A + (row0 + arow) * (long)K + (aq0 << 1);

  // prologue: stage tile 0
  {
    uint4 av = *(const uint4*)abase;
    *(uint4*)&Asm[0][arow * 16 + aq0] = av;
  }
  __syncthreads();

  for (int kt = 0; kt < ksteps; ++kt) {
    const int cur = kt & 1;
    uint4 avn;
    const bool more = (kt + 1) < ksteps;
    if (more) {
      // issue next A tile load early; prefetch next B-fragment stream
      avn = *(const uint4*)(abase + ((kt + 1) << 5));
      const unsigned int* bpn = Bp + (((long)(w << 1) * ksteps + (kt + 1)) * 32 + lane) * 8;
      __builtin_prefetch((const void*)bpn, 0, 3);
    }

    AFrag bfr[2];
    #pragma unroll
    for (int nt=0; nt<2; ++nt) {
      const int nT = (w << 1) + nt;
      const unsigned int* bp = Bp + (((long)nT * ksteps + kt) * 32 + lane) * 8;
      uint4 b0 = *(const uint4*)bp;
      uint4 b1 = *(const uint4*)(bp + 4);
      bfr[nt].u[0]=b0.x; bfr[nt].u[1]=b0.y; bfr[nt].u[2]=b0.z; bfr[nt].u[3]=b0.w;
      bfr[nt].u[4]=b1.x; bfr[nt].u[5]=b1.y; bfr[nt].u[6]=b1.z; bfr[nt].u[7]=b1.w;
    }
    #pragma unroll
    for (int mt=0; mt<4; ++mt) {
      AFrag af;
      #pragma unroll
      for (int p2=0; p2<8; ++p2) {
        int q = (p2 & 3) + (hi << 2) + ((p2 >> 2) << 3);
        af.u[p2] = Asm[cur][(mt * 16 + lr) * 16 + q];
      }
      acc[mt][0] = __builtin_amdgcn_wmma_f32_16x16x32_bf16(
          false, af.v, false, bfr[0].v, (short)0, acc[mt][0], false, false);
      acc[mt][1] = __builtin_amdgcn_wmma_f32_16x16x32_bf16(
          false, af.v, false, bfr[1].v, (short)0, acc[mt][1], false, false);
    }

    if (more) {
      *(uint4*)&Asm[cur ^ 1][arow * 16 + aq0] = avn;
      __syncthreads();        // publish next tile (different buffer: no WAR)
    }
  }

  #pragma unroll
  for (int mt=0; mt<4; ++mt) {
    #pragma unroll
    for (int nt=0; nt<2; ++nt) {
      const int col = (w << 5) + (nt << 4) + lr;
      #pragma unroll
      for (int r=0; r<8; ++r) {
        long row = row0 + mt * 16 + r + (hi << 3);
        float v = acc[mt][nt][r];
        if (bias)   v += bias[col];
        if (addsrc) v += addsrc[row * 256 + col];
        if (act)    v = silu_f(v);
        if (outf) outf[row * 256 + col] = v;
        if (outb) outb[row * 256 + col] = f2bf(v);
      }
    }
  }
}

// ---------------------------------------------------------------------------
// Small helper kernels
// ---------------------------------------------------------------------------
__global__ void f2bf_k(const float* __restrict__ src, unsigned short* __restrict__ dst, long n)
{
  long id = (long)blockIdx.x * 256 + threadIdx.x;
  if (id < n) dst[id] = f2bf(src[id]);
}

__global__ void copy_x_k(const float* __restrict__ src, float* __restrict__ dst)
{
  int id = blockIdx.x * 256 + threadIdx.x;
  if (id < ROWS * 3) dst[id] = src[id];
}

__global__ void din_k(const float* __restrict__ x0, float* __restrict__ din)
{
  int id = blockIdx.x * 256 + threadIdx.x;   // 0..EROWS-1
  int b = id >> 12, e = id & 4095, i = e >> 6, j = e & 63;
  const float* xi = x0 + ((size_t)b * 64 + i) * 3;
  const float* xj = x0 + ((size_t)b * 64 + j) * 3;
  float dx = xi[0]-xj[0], dy = xi[1]-xj[1], dz = xi[2]-xj[2];
  float d2 = dx*dx + dy*dy + dz*dz;
  din[id] = (i != j) ? sqrtf(fmaxf(d2, 1e-12f)) : 0.f;
}

__global__ void h0_k(const float* __restrict__ h_in, const float* __restrict__ t,
                     const float* __restrict__ ww, const float* __restrict__ wb,
                     float* __restrict__ h)
{
  int id = blockIdx.x * 256 + threadIdx.x;   // 0..ROWS*256-1
  int row = id >> 8, c = id & 255;
  float s = wb[c];
  #pragma unroll
  for (int k=0;k<5;k++) s += h_in[row*5 + k] * ww[k*256 + c];
  s += t[row] * ww[5*256 + c];
  h[id] = s;
}

// epre = silu( (i!=j ? P[i]+Q[j]+d2*wd2+din*wdin : 0) + bias ) -> bf16
__global__ __launch_bounds__(256) void edge_pre_k(
    const float* __restrict__ x, const float* __restrict__ P, const float* __restrict__ Q,
    const float* __restrict__ wd2, const float* __restrict__ wdin,
    const float* __restrict__ bias, const float* __restrict__ din,
    unsigned short* __restrict__ out)
{
  __shared__ float sd2;
  long eg = blockIdx.x;
  int c = threadIdx.x;
  int b = (int)(eg >> 12); int e = (int)(eg & 4095); int i = e >> 6, j = e & 63;
  if (c == 0) {
    const float* xi = x + ((size_t)b * 64 + i) * 3;
    const float* xj = x + ((size_t)b * 64 + j) * 3;
    float dx = xi[0]-xj[0], dy = xi[1]-xj[1], dz = xi[2]-xj[2];
    sd2 = fmaxf(dx*dx + dy*dy + dz*dz, 1e-12f);
  }
  __syncthreads();
  float pre = bias[c];
  if (i != j) {
    pre += P[((size_t)b*64 + i) * 256 + c] + Q[((size_t)b*64 + j) * 256 + c]
         + sd2 * wd2[c] + din[eg] * wdin[c];
  }
  out[eg * 256 + c] = f2bf(silu_f(pre));
}

// one wave32 per edge: out[e] = (sigmoid?) ( dot(bf16 row, wvec) + bias )
__global__ __launch_bounds__(256) void edge_dot_k(
    const unsigned short* __restrict__ Mb, const float* __restrict__ wvec,
    const float* __restrict__ bscalar, float* __restrict__ out, int do_sigmoid)
{
  int lane = threadIdx.x & 31;
  int wv   = threadIdx.x >> 5;
  long e = (long)blockIdx.x * 8 + wv;
  const unsigned short* row = Mb + e * 256 + lane * 8;
  float s = 0.f;
  #pragma unroll
  for (int q=0;q<8;q++) s += bf2f(row[q]) * wvec[lane*8 + q];
  #pragma unroll
  for (int off=16; off; off >>= 1) s += __shfl_xor(s, off, 32);
  if (lane == 0) {
    if (bscalar) s += *bscalar;
    if (do_sigmoid) s = sigm_f(s);
    out[e] = s;
  }
}

// em_agg[b,i,c] = sum_j gate[e] * m[e,c]  (edges of node i are contiguous)
__global__ __launch_bounds__(256) void seg_agg_k(
    const float* __restrict__ gate, const unsigned short* __restrict__ Mb,
    float* __restrict__ agg)
{
  int c = threadIdx.x;
  int bi = blockIdx.x;                 // (b*64+i)
  long ebase = (long)bi * 64;
  float acc = 0.f;
  for (int j=0;j<64;j++) {
    long e = ebase + j;
    acc += gate[e] * bf2f(Mb[e * 256 + c]);
  }
  agg[(size_t)bi * 256 + c] = acc;
}

__global__ void cat_k(const unsigned short* __restrict__ hb, const float* __restrict__ agg,
                      unsigned short* __restrict__ catb)
{
  int id = blockIdx.x * 256 + threadIdx.x;     // 0..ROWS*512-1
  int row = id >> 9, c = id & 511;
  catb[id] = (c < 256) ? hb[row*256 + c] : f2bf(agg[row*256 + (c - 256)]);
}

// x_new[b,i] = x[b,i] + sum_j diff/(d+1)*s  (cross-block safe: writes to xw)
__global__ __launch_bounds__(64) void x_update_k(
    const float* __restrict__ xr, const float* __restrict__ s, float* __restrict__ xw)
{
  __shared__ float red[64 * 3];
  int j = threadIdx.x;
  int bi = blockIdx.x;
  int b = bi >> 6, i = bi & 63;
  long e = (long)bi * 64 + j;
  const float* xi = xr + (size_t)bi * 3;
  const float* xj = xr + ((size_t)b * 64 + j) * 3;
  float dx = xi[0]-xj[0], dy = xi[1]-xj[1], dz = xi[2]-xj[2];
  if (j == i) { dx = 0.f; dy = 0.f; dz = 0.f; }
  float d = sqrtf(fmaxf(dx*dx + dy*dy + dz*dz, 1e-12f));
  float sc = s[e] / (d + 1.f);
  red[j*3+0] = dx*sc; red[j*3+1] = dy*sc; red[j*3+2] = dz*sc;
  __syncthreads();
  for (int off=32; off; off >>= 1) {
    if (j < off) {
      red[j*3+0] += red[(j+off)*3+0];
      red[j*3+1] += red[(j+off)*3+1];
      red[j*3+2] += red[(j+off)*3+2];
    }
    __syncthreads();
  }
  if (j < 3) xw[(size_t)bi * 3 + j] = xi[j] + red[j];
}

// out[...,0:3] = remove_mean(x - x_in)
__global__ __launch_bounds__(64) void final_x_k(
    const float* __restrict__ x, const float* __restrict__ x0, float* __restrict__ out)
{
  __shared__ float red[64 * 3];
  int i = threadIdx.x;
  int b = blockIdx.x;
  const float* xp  = x  + ((size_t)b * 64 + i) * 3;
  const float* x0p = x0 + ((size_t)b * 64 + i) * 3;
  float dv[3];
  #pragma unroll
  for (int a=0;a<3;a++) { dv[a] = xp[a] - x0p[a]; red[i*3+a] = dv[a]; }
  __syncthreads();
  for (int off=32; off; off >>= 1) {
    if (i < off) {
      red[i*3+0] += red[(i+off)*3+0];
      red[i*3+1] += red[(i+off)*3+1];
      red[i*3+2] += red[(i+off)*3+2];
    }
    __syncthreads();
  }
  #pragma unroll
  for (int a=0;a<3;a++)
    out[((size_t)b * 64 + i) * 8 + a] = dv[a] - red[a] * (1.f / 64.f);
}

// out[...,3:8] = (h @ wout_w + wout_b)[:, 0:5]
__global__ void final_h_k(const float* __restrict__ h, const float* __restrict__ ww,
                          const float* __restrict__ wb, float* __restrict__ out)
{
  int id = blockIdx.x * 256 + threadIdx.x;
  if (id >= ROWS * 5) return;
  int row = id / 5, c = id % 5;
  const float* hr = h + (size_t)row * 256;
  float s = wb[c];
  for (int k=0;k<256;k++) s += hr[k] * ww[k*6 + c];
  out[(size_t)row * 8 + 3 + c] = s;
}

// ---------------------------------------------------------------------------
extern "C" void kernel_launch(void* const* d_in, const int* in_sizes, int n_in,
                              void* d_out, int out_size, void* d_ws, size_t ws_size,
                              hipStream_t stream)
{
  const float* x_in   = (const float*)d_in[0];
  const float* h_in   = (const float*)d_in[1];
  const float* t_in   = (const float*)d_in[2];
  // d_in[3] node_mask (all ones), d_in[4] edge_mask (i!=j), d_in[5] edge_indices
  // -- structure is static (all-pairs), handled analytically.
  const float* win_w  = (const float*)d_in[6];
  const float* win_b  = (const float*)d_in[7];
  const float* wout_w = (const float*)d_in[8];
  const float* wout_b = (const float*)d_in[9];
  const float* we_w1  = (const float*)d_in[10];
  const float* we_b1  = (const float*)d_in[11];
  const float* we_w2  = (const float*)d_in[12];
  const float* we_b2  = (const float*)d_in[13];
  const float* attn_w = (const float*)d_in[14];
  const float* attn_b = (const float*)d_in[15];
  const float* wh_w1  = (const float*)d_in[16];
  const float* wh_b1  = (const float*)d_in[17];
  const float* wh_w2  = (const float*)d_in[18];
  const float* wh_b2  = (const float*)d_in[19];
  const float* wx_w1  = (const float*)d_in[20];
  const float* wx_b1  = (const float*)d_in[21];
  const float* wx_w2  = (const float*)d_in[22];
  const float* wx_b2  = (const float*)d_in[23];
  const float* wx_w3  = (const float*)d_in[24];
  float* out = (float*)d_out;

  // ---- workspace carve (aligned 256B) ----
  char* p = (char*)d_ws;
  auto alloc = [&](size_t bytes) -> char* {
    char* r = p; p += (bytes + 255) & ~(size_t)255; return r;
  };
  float*          xa     = (float*)alloc((size_t)ROWS * 3 * 4);
  float*          xb     = (float*)alloc((size_t)ROWS * 3 * 4);
  float*          h      = (float*)alloc((size_t)ROWS * 256 * 4);
  unsigned short* hb     = (unsigned short*)alloc((size_t)ROWS * 256 * 2);
  float*          dinb   = (float*)alloc((size_t)EROWS * 4);
  float*          P_e    = (float*)alloc((size_t)ROWS * 256 * 4);
  float*          Q_e    = (float*)alloc((size_t)ROWS * 256 * 4);
  float*          P_x    = (float*)alloc((size_t)ROWS * 256 * 4);
  float*          Q_x    = (float*)alloc((size_t)ROWS * 256 * 4);
  float*          em_agg = (float*)alloc((size_t)ROWS * 256 * 4);
  unsigned short* catb   = (unsigned short*)alloc((size_t)ROWS * 512 * 2);
  unsigned short* t1b    = (unsigned short*)alloc((size_t)ROWS * 256 * 2);
  float*          gate   = (float*)alloc((size_t)EROWS * 4);
  unsigned short* m1buf  = (unsigned short*)alloc((size_t)EROWS * 256 * 2);
  unsigned short* mbuf   = (unsigned short*)alloc((size_t)EROWS * 256 * 2);
  unsigned int*   WP     = (unsigned int*)alloc((size_t)LL * 294912 * 4);

  // per-layer packed-weight offsets (uint32 units)
  const size_t O_eA = 0, O_eB = 32768, O_e2 = 65536, O_h1 = 98304,
               O_h2 = 163840, O_xA = 196608, O_xB = 229376, O_x2 = 262144;
  auto wp = [&](int l) -> unsigned int* { return WP + (size_t)l * 294912; };

  // ---- pack all weights into WMMA B-fragment layout (stay hot in L2) ----
  for (int l = 0; l < LL; ++l) {
    pack_w_k<<<128, 256, 0, stream>>>(we_w1 + (size_t)l*514*256,              wp(l)+O_eA, 256);
    pack_w_k<<<128, 256, 0, stream>>>(we_w1 + (size_t)l*514*256 + 256*256,    wp(l)+O_eB, 256);
    pack_w_k<<<128, 256, 0, stream>>>(we_w2 + (size_t)l*65536,                wp(l)+O_e2, 256);
    pack_w_k<<<256, 256, 0, stream>>>(wh_w1 + (size_t)l*512*256,              wp(l)+O_h1, 512);
    pack_w_k<<<128, 256, 0, stream>>>(wh_w2 + (size_t)l*65536,                wp(l)+O_h2, 256);
    pack_w_k<<<128, 256, 0, stream>>>(wx_w1 + (size_t)l*514*256,              wp(l)+O_xA, 256);
    pack_w_k<<<128, 256, 0, stream>>>(wx_w1 + (size_t)l*514*256 + 256*256,    wp(l)+O_xB, 256);
    pack_w_k<<<128, 256, 0, stream>>>(wx_w2 + (size_t)l*65536,                wp(l)+O_x2, 256);
  }

  // ---- setup: x copy, fixed d_in, input embedding h0 ----
  copy_x_k<<<24, 256, 0, stream>>>(x_in, xa);
  din_k<<<EROWS/256, 256, 0, stream>>>(x_in, dinb);
  h0_k<<<ROWS, 256, 0, stream>>>(h_in, t_in, win_w, win_b, h);

  float* xr = xa;
  float* xw = xb;

  for (int l = 0; l < LL; ++l) {
    const size_t lw = (size_t)l * 514 * 256;

    // h -> bf16 for all GEMM A-operands this layer
    f2bf_k<<<ROWS, 256, 0, stream>>>(h, hb, (long)ROWS * 256);

    // node-level pre-GEMMs (the h_i / h_j factorization)
    wmma_gemm_k<<<ROWS/64, 256, 0, stream>>>(hb, wp(l)+O_eA, nullptr, nullptr, nullptr, P_e, 256, 0);
    wmma_gemm_k<<<ROWS/64, 256, 0, stream>>>(hb, wp(l)+O_eB, nullptr, nullptr, nullptr, Q_e, 256, 0);
    wmma_gemm_k<<<ROWS/64, 256, 0, stream>>>(hb, wp(l)+O_xA, nullptr, nullptr, nullptr, P_x, 256, 0);
    wmma_gemm_k<<<ROWS/64, 256, 0, stream>>>(hb, wp(l)+O_xB, nullptr, nullptr, nullptr, Q_x, 256, 0);

    // ---- edge-message path ----
    edge_pre_k<<<EROWS, 256, 0, stream>>>(xr, P_e, Q_e,
        we_w1 + lw + 512*256, we_w1 + lw + 513*256, we_b1 + (size_t)l*256, dinb, m1buf);
    wmma_gemm_k<<<EROWS/64, 256, 0, stream>>>(m1buf, wp(l)+O_e2,
        we_b2 + (size_t)l*256, nullptr, mbuf, nullptr, 256, 1);
    edge_dot_k<<<EROWS/8, 256, 0, stream>>>(mbuf, attn_w + (size_t)l*256,
        attn_b + l, gate, 1);
    seg_agg_k<<<ROWS, 256, 0, stream>>>(gate, mbuf, em_agg);

    // ---- node h update: h += MLP(concat(h, em_agg)) ----
    cat_k<<<ROWS*2, 256, 0, stream>>>(hb, em_agg, catb);
    wmma_gemm_k<<<ROWS/64, 256, 0, stream>>>(catb, wp(l)+O_h1,
        wh_b1 + (size_t)l*256, nullptr, t1b, nullptr, 512, 1);
    wmma_gemm_k<<<ROWS/64, 256, 0, stream>>>(t1b, wp(l)+O_h2,
        wh_b2 + (size_t)l*256, h, nullptr, h, 256, 0);

    // ---- coordinate path (uses OLD h via P_x/Q_x and OLD x) ----
    edge_pre_k<<<EROWS, 256, 0, stream>>>(xr, P_x, Q_x,
        wx_w1 + lw + 512*256, wx_w1 + lw + 513*256, wx_b1 + (size_t)l*256, dinb, m1buf);
    wmma_gemm_k<<<EROWS/64, 256, 0, stream>>>(m1buf, wp(l)+O_x2,
        wx_b2 + (size_t)l*256, nullptr, mbuf, nullptr, 256, 1);
    edge_dot_k<<<EROWS/8, 256, 0, stream>>>(mbuf, wx_w3 + (size_t)l*256,
        nullptr, gate, 0);                       // gate buffer reused as per-edge s
    x_update_k<<<ROWS, 64, 0, stream>>>(xr, gate, xw);

    float* tmp = xr; xr = xw; xw = tmp;          // ping-pong x
  }

  // ---- outputs: concat([remove_mean(x - x_in), (h@wout)[:, :5]], -1) ----
  final_x_k<<<BB, 64, 0, stream>>>(xr, x_in, out);
  final_h_k<<<(ROWS*5 + 255)/256, 256, 0, stream>>>(h, wout_w, wout_b, out);
}
